// GATNet_3470333575957
// MI455X (gfx1250) — matrix operand order repaired and checked by reading, not compiled, not run on hardware
//
#include <hip/hip_runtime.h>
#include <hip/hip_bf16.h>
#include <math.h>

// ---------------------------------------------------------------------------
// GAT 2-layer forward for MI455X (gfx1250, wave32).
//   layer1: H = X @ W1 (bf16 WMMA, f32 acc) -> attention softmax over edges
//           -> aggregate -> ELU
//   layer2: H2 = X2 @ W2 (bf16 WMMA) -> attention -> aggregate -> log_softmax
// Edge phase = 3 streaming passes (max / sum / normalize+scatter) with f32
// atomics; edge logits are recomputed per pass (cheaper than a 105MB array).
// GEMM fragments are loaded as 128-bit vectors: A is naturally two aligned
// 16B runs per lane; B is pre-packed into fragment-major order once per call.
// ---------------------------------------------------------------------------

typedef __bf16 bf16_t;
typedef __bf16 v16bf __attribute__((ext_vector_type(16)));
typedef float  v8f   __attribute__((ext_vector_type(8)));

union FragU {            // 32-byte WMMA operand fragment viewed as 2x uint4
    uint4 u[2];
    v16bf v;
};

#define NNODES   100000
#define NEDGES   3200000
#define E2       (NEDGES + NNODES)   // + self loops
#define IN_DIM   128
#define HID      64                  // H1 * F1
#define NHEADS   8
#define FDIM     8
#define NCLS     40
#define NCLSP    48                  // padded to 3x16 for WMMA tiles
#define NEG_SLOPE 0.2f

// ---------------------------------------------------------------- utilities
__device__ __forceinline__ void atomicMaxF(float* a, float v) {
    // monotonic float-as-int trick: works for mixed signs with -inf init
    if (v >= 0.0f) atomicMax((int*)a, __float_as_int(v));
    else           atomicMin((unsigned int*)a, __float_as_uint(v));
}

__device__ __forceinline__ void edge_ids(const int* __restrict__ adj, int e,
                                         int& s, int& d) {
    if (e < NEDGES) { s = adj[e]; d = adj[NEDGES + e]; }
    else            { s = d = e - NEDGES; }   // appended self loop
}

__device__ __forceinline__ float leaky(float x) {
    return x > 0.0f ? x : NEG_SLOPE * x;
}

// ------------------------------------------------------------- init/convert
__global__ void k_f32_to_bf16(const float* __restrict__ s,
                              bf16_t* __restrict__ d, int n) {
    int i = blockIdx.x * blockDim.x + threadIdx.x;
    if (i < n) d[i] = (bf16_t)s[i];
}

// Pack B [K x NReal] f32 into bf16 WMMA-fragment-major order for [K x N]:
//   out[(((ct*(K/32) + kt)*32 + lane)*16 + j] = B[kt*32 + (j<8 ? hi*8+j : 16+hi*8+j-8)][ct*16+n]
// so each lane's 32-byte fragment is contiguous (two b128 loads in the GEMM).
__global__ void k_pack_B(const float* __restrict__ W, bf16_t* __restrict__ out,
                         int K, int N, int NReal) {
    int i = blockIdx.x * blockDim.x + threadIdx.x;
    if (i >= K * N) return;
    int j   = i & 15;
    int l   = (i >> 4) & 31;
    int kt  = (i >> 9) % (K / 32);
    int ct  = (i >> 9) / (K / 32);
    int n   = l & 15, hi = l >> 4;
    int row = kt * 32 + (j < 8 ? hi * 8 + j : 16 + hi * 8 + (j - 8));
    int col = ct * 16 + n;
    out[i] = (bf16_t)(col < NReal ? W[row * NReal + col] : 0.0f);
}

__global__ void k_fill(float* __restrict__ p, float v, int n) {
    int i = blockIdx.x * blockDim.x + threadIdx.x;
    if (i < n) p[i] = v;
}

// ------------------------------------------------------------- WMMA GEMM
// C[M x N] = A[M x K](bf16, row major) * B(bf16, pre-packed), f32 accumulate.
// One wave per 16x16 tile; K,N compile-time so the K loop fully unrolls.
// Fragment layouts per CDNA5 ISA 7.12.2 (wave32).
template <int K, int N>
__global__ void k_wmma_gemm(const bf16_t* __restrict__ A,
                            const bf16_t* __restrict__ Bpack,
                            float* __restrict__ C, int M) {
    int wid = blockIdx.x * (blockDim.x >> 5) + (threadIdx.x >> 5);
    const int ncol = N / 16;
    int ntiles = (M / 16) * ncol;
    if (wid >= ntiles) return;                 // wave-uniform: EXEC stays full

    int rt = wid / ncol, ct = wid % ncol;
    int lane = threadIdx.x & 31;
    int mr = lane & 15;        // A row / C column within tile
    int hi = lane >> 4;        // lane half selector

    const bf16_t* Arow = A + (size_t)(rt * 16 + mr) * K;
    const bf16_t* Bpk  = Bpack + ((size_t)ct * (K / 32) * 32 + lane) * 16;

    v8f acc = {};
#pragma unroll
    for (int kt = 0; kt < K / 32; ++kt) {
        FragU fa, fb;
        // A: 16-bit 16x32 layout -> two aligned 16B runs per lane
        const uint4* ap = (const uint4*)(Arow + kt * 32 + hi * 8);
        fa.u[0] = ap[0];       // K = k0 + hi*8      .. +7
        fa.u[1] = ap[2];       // K = k0 + 16 + hi*8 .. +7
        // B: pre-packed fragment, 32 contiguous bytes per lane
        const uint4* bp = (const uint4*)(Bpk + (size_t)kt * 32 * 16);
        fb.u[0] = bp[0];
        fb.u[1] = bp[1];
        acc = __builtin_amdgcn_wmma_f32_16x16x32_bf16(
                  false, fa.v, false, fb.v, (short)0, acc, false, false);
    }

    float* Cp = C + (size_t)(rt * 16) * N + ct * 16 + mr;
#pragma unroll
    for (int r = 0; r < 8; ++r)                // C: VGPR r -> row r + 8*hi
        Cp[(size_t)(r + 8 * hi) * N] = acc[r];
}

// ------------------------------------------------- per-node attention coeffs
// es[n,h] = sum_f H[n, h*fdim+f] * a_s[h,f]   (ed likewise)
__global__ void k_attn(const float* __restrict__ Hf,
                       const float* __restrict__ a_s,
                       const float* __restrict__ a_d,
                       float* __restrict__ es, float* __restrict__ ed,
                       int heads, int fdim, int ldh) {
    int i = blockIdx.x * blockDim.x + threadIdx.x;
    if (i >= NNODES * heads) return;
    int node = i / heads, h = i % heads;
    const float* hp = Hf + (size_t)node * ldh + h * fdim;
    float s = 0.0f, d = 0.0f;
    for (int f = 0; f < fdim; ++f) {
        s += hp[f] * a_s[h * fdim + f];
        d += hp[f] * a_d[h * fdim + f];
    }
    es[i] = s; ed[i] = d;
}

// ------------------------------------------------------ edge pass 1: seg max
__global__ void k_edge_max(const int* __restrict__ adj,
                           const float* __restrict__ es,
                           const float* __restrict__ ed,
                           float* __restrict__ m, int heads) {
    int i = blockIdx.x * blockDim.x + threadIdx.x;
    if (i >= E2 * heads) return;
    int e = i / heads, h = i % heads;
    int s, d; edge_ids(adj, e, s, d);
    float x = leaky(es[s * heads + h] + ed[d * heads + h]);
    atomicMaxF(&m[d * heads + h], x);
}

// ------------------------------------------------------ edge pass 2: seg sum
__global__ void k_edge_sum(const int* __restrict__ adj,
                           const float* __restrict__ es,
                           const float* __restrict__ ed,
                           const float* __restrict__ m,
                           float* __restrict__ sums, int heads) {
    int i = blockIdx.x * blockDim.x + threadIdx.x;
    if (i >= E2 * heads) return;
    int e = i / heads, h = i % heads;
    int s, d; edge_ids(adj, e, s, d);
    float x = leaky(es[s * heads + h] + ed[d * heads + h]);
    atomicAdd(&sums[d * heads + h], __expf(x - m[d * heads + h]));
}

// ----------------------------------------- edge pass 3: normalize + scatter
// layer 1: 64 threads per edge (j = h*8+f)
__global__ void k_edge_agg1(const int* __restrict__ adj,
                            const float* __restrict__ es,
                            const float* __restrict__ ed,
                            const float* __restrict__ m,
                            const float* __restrict__ sums,
                            const float* __restrict__ Hf,
                            float* __restrict__ out) {
    int i = blockIdx.x * blockDim.x + threadIdx.x;
    if (i >= E2 * HID) return;
    int e = i >> 6, j = i & 63, h = j >> 3;
    int s, d; edge_ids(adj, e, s, d);
    float x = leaky(es[s * NHEADS + h] + ed[d * NHEADS + h]);
    float alpha = __expf(x - m[d * NHEADS + h]) / sums[d * NHEADS + h];
    atomicAdd(&out[(size_t)d * HID + j], alpha * Hf[(size_t)s * HID + j]);
}

// layer 2: 40 threads per edge, single head, h2 has leading dim 48
__global__ void k_edge_agg2(const int* __restrict__ adj,
                            const float* __restrict__ es,
                            const float* __restrict__ ed,
                            const float* __restrict__ m,
                            const float* __restrict__ sums,
                            const float* __restrict__ h2,
                            float* __restrict__ out) {
    int i = blockIdx.x * blockDim.x + threadIdx.x;
    if (i >= E2 * NCLS) return;
    int e = i / NCLS, j = i % NCLS;
    int s, d; edge_ids(adj, e, s, d);
    float x = leaky(es[s] + ed[d]);
    float alpha = __expf(x - m[d]) / sums[d];
    atomicAdd(&out[(size_t)d * NCLS + j], alpha * h2[(size_t)s * NCLSP + j]);
}

// ---------------------------------------------------------------- ELU->bf16
__global__ void k_elu_bf16(const float* __restrict__ in,
                           bf16_t* __restrict__ out, int n) {
    int i = blockIdx.x * blockDim.x + threadIdx.x;
    if (i < n) {
        float v = in[i];
        v = v > 0.0f ? v : (__expf(v) - 1.0f);
        out[i] = (bf16_t)v;
    }
}

// --------------------------------------------------------------- logsoftmax
__global__ void k_lsm(const float* __restrict__ in, float* __restrict__ out) {
    int i = blockIdx.x * blockDim.x + threadIdx.x;
    if (i >= NNODES) return;
    const float* r = in + (size_t)i * NCLS;
    float mx = r[0];
    for (int c = 1; c < NCLS; ++c) mx = fmaxf(mx, r[c]);
    float s = 0.0f;
    for (int c = 0; c < NCLS; ++c) s += __expf(r[c] - mx);
    float ls = __logf(s);
    float* o = out + (size_t)i * NCLS;
    for (int c = 0; c < NCLS; ++c) o[c] = r[c] - mx - ls;
}

// ---------------------------------------------------------------------------
static inline char* bump(char*& p, size_t bytes) {
    char* r = p;
    p += (bytes + 255) & ~(size_t)255;
    return r;
}
static inline unsigned cdiv(long long a, long long b) {
    return (unsigned)((a + b - 1) / b);
}

extern "C" void kernel_launch(void* const* d_in, const int* in_sizes, int n_in,
                              void* d_out, int out_size, void* d_ws, size_t ws_size,
                              hipStream_t stream) {
    const float* X    = (const float*)d_in[0];   // [N,128]
    const int*   adj  = (const int*)d_in[1];     // [2,E]
    const float* W1   = (const float*)d_in[2];   // [128,64]
    const float* a1s  = (const float*)d_in[3];   // [8,8]
    const float* a1d  = (const float*)d_in[4];
    const float* W2   = (const float*)d_in[5];   // [64,40]
    const float* a2s  = (const float*)d_in[6];   // [1,40]
    const float* a2d  = (const float*)d_in[7];
    float* outp = (float*)d_out;                 // [N,40]

    char* ws = (char*)d_ws;
    bf16_t* Xb   = (bf16_t*)bump(ws, (size_t)NNODES * IN_DIM * 2);
    bf16_t* W1p  = (bf16_t*)bump(ws, (size_t)IN_DIM * HID * 2);     // packed
    bf16_t* W2p  = (bf16_t*)bump(ws, (size_t)HID * NCLSP * 2);      // packed
    float*  Hf   = (float*) bump(ws, (size_t)NNODES * HID * 4);
    float*  e1s  = (float*) bump(ws, (size_t)NNODES * NHEADS * 4);
    float*  e1d  = (float*) bump(ws, (size_t)NNODES * NHEADS * 4);
    float*  m1   = (float*) bump(ws, (size_t)NNODES * NHEADS * 4);
    float*  s1   = (float*) bump(ws, (size_t)NNODES * NHEADS * 4);
    float*  out1 = (float*) bump(ws, (size_t)NNODES * HID * 4);
    bf16_t* X2b  = (bf16_t*)bump(ws, (size_t)NNODES * HID * 2);
    float*  h2   = (float*) bump(ws, (size_t)NNODES * NCLSP * 4);
    float*  e2s  = (float*) bump(ws, (size_t)NNODES * 4);
    float*  e2d  = (float*) bump(ws, (size_t)NNODES * 4);
    float*  m2   = (float*) bump(ws, (size_t)NNODES * 4);
    float*  s2   = (float*) bump(ws, (size_t)NNODES * 4);
    float*  out2 = (float*) bump(ws, (size_t)NNODES * NCLS * 4);

    const int T = 256;
    const float NEG_INF = -__builtin_huge_valf();

    // ---- conversions + accumulator init (re-done every call: deterministic)
    k_f32_to_bf16<<<cdiv((long long)NNODES * IN_DIM, T), T, 0, stream>>>(X, Xb, NNODES * IN_DIM);
    k_pack_B<<<cdiv(IN_DIM * HID, T), T, 0, stream>>>(W1, W1p, IN_DIM, HID, HID);
    k_pack_B<<<cdiv(HID * NCLSP, T), T, 0, stream>>>(W2, W2p, HID, NCLSP, NCLS);
    k_fill<<<cdiv(NNODES * NHEADS, T), T, 0, stream>>>(m1, NEG_INF, NNODES * NHEADS);
    k_fill<<<cdiv(NNODES * NHEADS, T), T, 0, stream>>>(s1, 0.0f, NNODES * NHEADS);
    k_fill<<<cdiv((long long)NNODES * HID, T), T, 0, stream>>>(out1, 0.0f, NNODES * HID);
    k_fill<<<cdiv(NNODES, T), T, 0, stream>>>(m2, NEG_INF, NNODES);
    k_fill<<<cdiv(NNODES, T), T, 0, stream>>>(s2, 0.0f, NNODES);
    k_fill<<<cdiv((long long)NNODES * NCLS, T), T, 0, stream>>>(out2, 0.0f, NNODES * NCLS);

    // ---- layer 1 ----
    {   // H = X @ W1 : tiles = (100000/16)*(64/16) = 25000 waves
        unsigned waves = (NNODES / 16) * (HID / 16);
        k_wmma_gemm<IN_DIM, HID><<<cdiv(waves, T / 32), T, 0, stream>>>(Xb, W1p, Hf, NNODES);
    }
    k_attn<<<cdiv((long long)NNODES * NHEADS, T), T, 0, stream>>>(Hf, a1s, a1d, e1s, e1d, NHEADS, FDIM, HID);
    k_edge_max<<<cdiv((long long)E2 * NHEADS, T), T, 0, stream>>>(adj, e1s, e1d, m1, NHEADS);
    k_edge_sum<<<cdiv((long long)E2 * NHEADS, T), T, 0, stream>>>(adj, e1s, e1d, m1, s1, NHEADS);
    k_edge_agg1<<<cdiv((long long)E2 * HID, T), T, 0, stream>>>(adj, e1s, e1d, m1, s1, Hf, out1);
    k_elu_bf16<<<cdiv((long long)NNODES * HID, T), T, 0, stream>>>(out1, X2b, NNODES * HID);

    // ---- layer 2 ----
    {   // H2 = X2 @ W2pad : tiles = (100000/16)*(48/16) = 18750 waves
        unsigned waves = (NNODES / 16) * (NCLSP / 16);
        k_wmma_gemm<HID, NCLSP><<<cdiv(waves, T / 32), T, 0, stream>>>(X2b, W2p, h2, NNODES);
    }
    k_attn<<<cdiv(NNODES, T), T, 0, stream>>>(h2, a2s, a2d, e2s, e2d, 1, NCLS, NCLSP);
    k_edge_max<<<cdiv((long long)E2, T), T, 0, stream>>>(adj, e2s, e2d, m2, 1);
    k_edge_sum<<<cdiv((long long)E2, T), T, 0, stream>>>(adj, e2s, e2d, m2, s2, 1);
    k_edge_agg2<<<cdiv((long long)E2 * NCLS, T), T, 0, stream>>>(adj, e2s, e2d, m2, s2, h2, out2);

    // ---- log_softmax ----
    k_lsm<<<cdiv(NNODES, T), T, 0, stream>>>(out2, outp);
}
